// Sampler_ASGCN_4698694222356
// MI455X (gfx1250) — matrix operand order repaired and checked by reading, not compiled, not run on hardware
//
#include <hip/hip_runtime.h>
#include <hip/hip_bf16.h>
#include <stdint.h>

// ---------------------------------------------------------------------------
// ASGCN sampler forward for MI455X (gfx1250, wave32).
// Outputs (flat, float32): sampled_X0 [256,128] | support0 [512,256] |
//                          support1 [1024,512] | loss [1]
// ---------------------------------------------------------------------------

#define NN 50000
#define DD 128
#define BB 1024
#define KK 32
#define L0 256
#define L1 512

typedef __attribute__((ext_vector_type(16))) _Float16 v16h;
typedef __attribute__((ext_vector_type(8)))  float    v8f;

// ----------------------------- helpers -------------------------------------

__device__ __forceinline__ float u01_hash(unsigned long long seed, unsigned int j) {
    unsigned long long z = seed + 0x9E3779B97F4A7C15ULL * (unsigned long long)(j + 1u);
    z = (z ^ (z >> 30)) * 0xBF58476D1CE4E5B9ULL;
    z = (z ^ (z >> 27)) * 0x94D049BB133111EBULL;
    z ^= (z >> 31);
    return (float)(z >> 40) * (1.0f / 16777216.0f);   // 24-bit mantissa uniform [0,1)
}

// ----------------------------- kernels -------------------------------------

__global__ void zero_kernel(float* __restrict__ p, int n) {
    int t = blockIdx.x * blockDim.x + threadIdx.x;
    if (t < n) p[t] = 0.0f;
}

// fw1[n] = features[n,:] . w1   fw2[n] = features[n,:] . w2
// One wave per 16-row tile; D=128 consumed as 4 chunks of K=32 through
// v_wmma_f32_16x16x32_f16. B carries w1 in column 0, w2 in column 1.
__global__ void fw_wmma_kernel(const float* __restrict__ features,
                               const float* __restrict__ w1,
                               const float* __restrict__ w2,
                               float* __restrict__ fw1,
                               float* __restrict__ fw2) {
    const int wave = blockIdx.x * (blockDim.x >> 5) + (threadIdx.x >> 5);
    const int NT = NN / 16;                 // 3125 tiles, exact
    if (wave >= NT) return;                 // whole-wave uniform exit (EXEC all-1 inside)
    const int lane = threadIdx.x & 31;
    const int mrow = lane & 15;
    const int hi   = lane >> 4;             // 0: lanes 0-15, 1: lanes 16-31
    const int row  = wave * 16 + mrow;

    v8f acc = {};
    for (int c = 0; c < 4; ++c) {
        const int kc = c * 32;
        // A (16x32 f16): lane<16 e:0..7 -> K=kc+e, e:8..15 -> K=kc+16+e-8
        //                lane>=16 shifted by +8 (ISA 7.12.2 layout)
        v16h a;
        const float* fr = features + (size_t)row * DD + kc + hi * 8;
        #pragma unroll
        for (int e = 0; e < 8; ++e) a[e]     = (_Float16)fr[e];
        #pragma unroll
        for (int e = 0; e < 8; ++e) a[8 + e] = (_Float16)fr[16 + e];
        // B (32x16 f16): col = lane&15; same K layout as A (e:0..7 -> K=kc+8*hi+e,
        // e:8..15 -> K=kc+16+8*hi+e-8). Only columns 0 (w1) and 1 (w2).
        v16h b = (v16h)(_Float16)0.0f;
        const int col = lane & 15;
        if (col < 2) {
            const float* wp = (col == 0 ? w1 : w2) + kc + hi * 8;
            #pragma unroll
            for (int e = 0; e < 8; ++e) b[e] = (_Float16)wp[e];
            #pragma unroll
            for (int e = 0; e < 8; ++e) b[8 + e] = (_Float16)wp[16 + e];
        }
        acc = __builtin_amdgcn_wmma_f32_16x16x32_f16(
                  false, a, false, b, (short)0, acc, false, false);
    }
    // C layout: lane<16 -> C[M=g][N=lane], lane>=16 -> C[M=8+g][N=lane-16].
    // Column 0 = fw1, column 1 = fw2; each lane holds 8 consecutive rows.
    const int n     = lane & 15;
    const int mbase = wave * 16 + hi * 8;
    if (n == 0) {
        #pragma unroll
        for (int g = 0; g < 8; ++g) fw1[mbase + g] = acc[g];
    } else if (n == 1) {
        #pragma unroll
        for (int g = 0; g < 8; ++g) fw2[mbase + g] = acc[g];
    }
}

__global__ void scatter_colsum_kernel(const int* __restrict__ v_idx, int b,
                                      const int* __restrict__ nbr_idx,
                                      const float* __restrict__ nbr_val,
                                      float* __restrict__ colsum) {
    int t = blockIdx.x * blockDim.x + threadIdx.x;
    if (t >= b * KK) return;
    int i = t / KK, k = t - i * KK;
    int vi = v_idx[i];
    atomicAdd(&colsum[nbr_idx[(size_t)vi * KK + k]], nbr_val[(size_t)vi * KK + k]);
}

__global__ void count_pos_kernel(const float* __restrict__ colsum,
                                 float* __restrict__ out) {
    __shared__ float red[256];
    int t = threadIdx.x;
    float acc = 0.0f;
    for (int n = blockIdx.x * blockDim.x + t; n < NN; n += gridDim.x * blockDim.x)
        acc += (colsum[n] > 0.0f) ? 1.0f : 0.0f;
    red[t] = acc; __syncthreads();
    for (int s = 128; s > 0; s >>= 1) { if (t < s) red[t] += red[t + s]; __syncthreads(); }
    if (t == 0) atomicAdd(out, red[0]);
}

// p1[node] += val * relu(fw1[v_i] + fw2[node] + 1) / n_neis
__global__ void scatter_p1_kernel(const int* __restrict__ v_idx, int b,
                                  const int* __restrict__ nbr_idx,
                                  const float* __restrict__ nbr_val,
                                  const float* __restrict__ fw1,
                                  const float* __restrict__ fw2,
                                  const float* __restrict__ nneis_p,
                                  float* __restrict__ p1) {
    int t = blockIdx.x * blockDim.x + threadIdx.x;
    if (t >= b * KK) return;
    int i = t / KK, k = t - i * KK;
    int vi   = v_idx[i];
    int node = nbr_idx[(size_t)vi * KK + k];
    float att = fw1[vi] + fw2[node] + 1.0f;
    att = fmaxf(att, 0.0f) / nneis_p[0];
    atomicAdd(&p1[node], nbr_val[(size_t)vi * KK + k] * att);
}

// Inclusive prefix sum over p1 (single 1024-thread block, 49 chunk passes).
__global__ void scan_kernel(const float* __restrict__ p1, float* __restrict__ cdf,
                            float* __restrict__ total) {
    __shared__ float s[1024];
    __shared__ float carry;
    const int t = threadIdx.x;
    if (t == 0) carry = 0.0f;
    __syncthreads();
    for (int base = 0; base < NN; base += 1024) {
        float x = (base + t < NN) ? p1[base + t] : 0.0f;
        s[t] = x;
        __syncthreads();
        for (int off = 1; off < 1024; off <<= 1) {
            float v = (t >= off) ? s[t - off] : 0.0f;
            __syncthreads();
            s[t] += v;
            __syncthreads();
        }
        if (base + t < NN) cdf[base + t] = carry + s[t];
        __syncthreads();
        if (t == 0) carry += s[1023];
        __syncthreads();
    }
    if (t == 0) total[0] = carry;
}

// Inverse-CDF categorical sampling (deterministic hash RNG).
__global__ void sample_kernel(const float* __restrict__ cdf,
                              const float* __restrict__ total_p,
                              int* __restrict__ out_idx, int count,
                              unsigned long long seed) {
    int j = blockIdx.x * blockDim.x + threadIdx.x;
    if (j >= count) return;
    float x = u01_hash(seed, (unsigned)j) * total_p[0];
    int lo = 0, hi = NN;
    while (lo < hi) {
        int mid = (lo + hi) >> 1;
        if (cdf[mid] > x) hi = mid; else lo = mid + 1;
    }
    out_idx[j] = (lo < NN) ? lo : (NN - 1);
}

// sup[i,j] = support[i, sampled[j]] / (p1[sampled[j]] * out_size)
__global__ void sup_kernel(const int* __restrict__ v_idx,
                           const int* __restrict__ nbr_idx,
                           const float* __restrict__ nbr_val,
                           const int* __restrict__ sampled,
                           const float* __restrict__ p1,
                           float* __restrict__ out, int out_size) {
    __shared__ int   snode[KK];
    __shared__ float sval[KK];
    const int i  = blockIdx.x;
    const int vi = v_idx[i];
    if (threadIdx.x < KK) {
        snode[threadIdx.x] = nbr_idx[(size_t)vi * KK + threadIdx.x];
        sval[threadIdx.x]  = nbr_val[(size_t)vi * KK + threadIdx.x];
    }
    __syncthreads();
    for (int j = threadIdx.x; j < out_size; j += blockDim.x) {
        const int nj = sampled[j];
        float acc = 0.0f;
        #pragma unroll
        for (int k = 0; k < KK; ++k) acc += (snode[k] == nj) ? sval[k] : 0.0f;
        const float pj = p1[nj];
        out[(size_t)i * out_size + j] = (pj > 0.0f) ? acc / (pj * (float)out_size) : 0.0f;
    }
}

__global__ void gather_rows_kernel(const int* __restrict__ idx,
                                   const float* __restrict__ features,
                                   float* __restrict__ out, int rows) {
    int t = blockIdx.x * blockDim.x + threadIdx.x;
    if (t >= rows * DD) return;
    int r = t / DD, d = t - r * DD;
    out[t] = features[(size_t)idx[r] * DD + d];
}

// means[d] = sum over masked rows of features[n,d]   (mask = colsum>0)
__global__ void means_kernel(const float* __restrict__ features,
                             const float* __restrict__ colsum,
                             float* __restrict__ means) {
    const int d = threadIdx.x;                 // blockDim == 128
    float acc = 0.0f;
    for (int n = blockIdx.x; n < NN; n += gridDim.x) {
        if (n + (int)gridDim.x < NN)
            __builtin_prefetch(&features[(size_t)(n + gridDim.x) * DD + d]);
        if (colsum[n] > 0.0f) acc += features[(size_t)n * DD + d];
    }
    atomicAdd(&means[d], acc);
}

// loss_acc += sum_n p_u[n] * sum_d (f[n,d]-means[d])^2
__global__ void loss_kernel(const float* __restrict__ features,
                            const float* __restrict__ p1,
                            const float* __restrict__ total_p,
                            const float* __restrict__ means,
                            float* __restrict__ loss_acc) {
    __shared__ float red[128];
    const int d = threadIdx.x;                 // blockDim == 128
    const float m = means[d];
    const float T = total_p[0];
    const float invT = (T > 0.0f) ? 1.0f / T : 0.0f;
    float acc = 0.0f;
    for (int n = blockIdx.x; n < NN; n += gridDim.x) {
        if (n + (int)gridDim.x < NN)
            __builtin_prefetch(&features[(size_t)(n + gridDim.x) * DD + d]);
        const float w = p1[n] * invT;
        const float f = features[(size_t)n * DD + d] - m;
        acc = fmaf(f * f, w, acc);
    }
    red[d] = acc; __syncthreads();
    for (int s = 64; s > 0; s >>= 1) { if (d < s) red[d] += red[d + s]; __syncthreads(); }
    if (d == 0) atomicAdd(loss_acc, red[0]);
}

__global__ void finalize_loss_kernel(const float* __restrict__ loss_acc,
                                     float* __restrict__ out_loss) {
    if (threadIdx.x == 0 && blockIdx.x == 0)
        out_loss[0] = loss_acc[0] / (float)DD;
}

// ----------------------------- launch --------------------------------------

extern "C" void kernel_launch(void* const* d_in, const int* in_sizes, int n_in,
                              void* d_out, int out_size, void* d_ws, size_t ws_size,
                              hipStream_t stream) {
    const int*   v        = (const int*)  d_in[0];
    const float* features = (const float*)d_in[1];
    const float* w1       = (const float*)d_in[2];
    const float* w2       = (const float*)d_in[3];
    const int*   nbr_idx  = (const int*)  d_in[4];
    const float* nbr_val  = (const float*)d_in[5];
    float* out = (float*)d_out;

    // Workspace layout (floats): fw1|fw2|colsum1|p1_1|cdf1|colsum0|p1_0|cdf0|means|scal
    float* ws      = (float*)d_ws;
    float* fw1     = ws;
    float* fw2     = ws + 1 * NN;
    float* colsum1 = ws + 2 * NN;
    float* p1_1    = ws + 3 * NN;
    float* cdf1    = ws + 4 * NN;
    float* colsum0 = ws + 5 * NN;
    float* p1_0    = ws + 6 * NN;
    float* cdf0    = ws + 7 * NN;
    float* means   = ws + 8 * NN;
    float* scal    = ws + 8 * NN + DD;   // [0]=nneis1 [1]=nneis0 [2]=T1 [3]=T0 [4]=loss
    int*   sampled1 = (int*)(scal + 8);
    int*   sampled0 = sampled1 + L1;

    // Output offsets: X0 | support0 | support1 | loss
    float* out_x0   = out;
    float* out_sup0 = out + (size_t)L0 * DD;                       // 32768
    float* out_sup1 = out_sup0 + (size_t)L1 * L0;                  // 163840
    float* out_loss = out_sup1 + (size_t)BB * L1;                  // 688128

    // 0) zero accumulators (colsum/p1/cdf pads/means/scal)
    {
        int n = 6 * NN + DD + 8;
        zero_kernel<<<(n + 255) / 256, 256, 0, stream>>>(colsum1, n);
    }
    // 1) fw1/fw2 via WMMA (3125 tiles, 4 waves/block)
    fw_wmma_kernel<<<(NN / 16 + 3) / 4, 128, 0, stream>>>(features, w1, w2, fw1, fw2);

    // ---- layer i=1: v_idx = v (b=1024), out_size = 512 ----
    scatter_colsum_kernel<<<(BB * KK + 255) / 256, 256, 0, stream>>>(v, BB, nbr_idx, nbr_val, colsum1);
    count_pos_kernel<<<64, 256, 0, stream>>>(colsum1, &scal[0]);
    scatter_p1_kernel<<<(BB * KK + 255) / 256, 256, 0, stream>>>(v, BB, nbr_idx, nbr_val,
                                                                 fw1, fw2, &scal[0], p1_1);
    scan_kernel<<<1, 1024, 0, stream>>>(p1_1, cdf1, &scal[2]);
    sample_kernel<<<(L1 + 255) / 256, 256, 0, stream>>>(cdf1, &scal[2], sampled1, L1,
                                                        0x5DEECE66D1ULL);
    sup_kernel<<<BB, 256, 0, stream>>>(v, nbr_idx, nbr_val, sampled1, p1_1, out_sup1, L1);

    // ---- layer i=0: v_idx = sampled1 (b=512), out_size = 256 ----
    scatter_colsum_kernel<<<(L1 * KK + 255) / 256, 256, 0, stream>>>(sampled1, L1, nbr_idx, nbr_val, colsum0);
    count_pos_kernel<<<64, 256, 0, stream>>>(colsum0, &scal[1]);
    scatter_p1_kernel<<<(L1 * KK + 255) / 256, 256, 0, stream>>>(sampled1, L1, nbr_idx, nbr_val,
                                                                 fw1, fw2, &scal[1], p1_0);
    scan_kernel<<<1, 1024, 0, stream>>>(p1_0, cdf0, &scal[3]);
    sample_kernel<<<(L0 + 255) / 256, 256, 0, stream>>>(cdf0, &scal[3], sampled0, L0,
                                                        0xB5297A4D2ULL);
    sup_kernel<<<L1, 256, 0, stream>>>(sampled1, nbr_idx, nbr_val, sampled0, p1_0, out_sup0, L0);

    // ---- epilogue ----
    gather_rows_kernel<<<(L0 * DD + 255) / 256, 256, 0, stream>>>(sampled0, features, out_x0, L0);
    means_kernel<<<256, 128, 0, stream>>>(features, colsum1, means);
    loss_kernel<<<256, 128, 0, stream>>>(features, p1_1, &scal[2], means, &scal[4]);
    finalize_loss_kernel<<<1, 32, 0, stream>>>(&scal[4], out_loss);
}
